// Heat2D_81612968559185
// MI455X (gfx1250) — matrix operand order repaired and checked by reading, not compiled, not run on hardware
//
#include <hip/hip_runtime.h>

// ---------------- CDNA5 (gfx1250) types ----------------
typedef __attribute__((ext_vector_type(16))) __bf16 v16bf;
typedef __attribute__((ext_vector_type(8)))  float  v8f;
typedef __attribute__((ext_vector_type(4)))  int    v4i;

struct __attribute__((aligned(16))) U4 { unsigned int a, b, c, d; };
struct __attribute__((aligned(16))) F4 { float x, y, z, w; };

union FragAB { v16bf v; unsigned short s[16]; U4 q[2]; };
union FragC  { v8f   v; float f[8]; };

__device__ __forceinline__ unsigned short f2bf(float x) {
  unsigned int u = __float_as_uint(x);
  u += 0x7fffu + ((u >> 16) & 1u);          // round-to-nearest-even
  return (unsigned short)(u >> 16);
}
__device__ __forceinline__ float bf2f(unsigned short h) {
  return __uint_as_float(((unsigned int)h) << 16);
}

// ---------------- async global->LDS staging (ASYNCcnt path), guarded ----------------
#if __has_builtin(__builtin_amdgcn_global_load_async_to_lds_b128) && \
    __has_builtin(__builtin_amdgcn_s_wait_asynccnt)
#define ASYNC_OK 1
#else
#define ASYNC_OK 0
#endif

__device__ __forceinline__ void async_copy_b128(const void* g, void* l) {
#if ASYNC_OK
  void* gg = const_cast<void*>(g);
  __builtin_amdgcn_global_load_async_to_lds_b128(
      (__attribute__((address_space(1))) v4i*)gg,
      (__attribute__((address_space(3))) v4i*)l, 0, 0);
#else
  *(U4*)l = *(const U4*)g;
#endif
}
__device__ __forceinline__ void async_wait_all() {
#if ASYNC_OK
  __builtin_amdgcn_s_wait_asynccnt(0);
#endif
}

#define PI_F 3.14159265358979323846f

// Problem constants
#define BB   64
#define CC   192
#define C2   384
#define HH   56
#define HW   3136           // 56*56
#define MM   200704         // B*HW

// ---------------- prep: DCT matrix (64x64 bf16, zero-padded) + decay factors ----------------
__global__ void prep_const_kernel(unsigned short* __restrict__ Dp, float* __restrict__ dfac) {
  int i = blockIdx.x * blockDim.x + threadIdx.x;
  if (i < 4096) {
    int m = i >> 6, w = i & 63;
    float v = 0.f;
    if (m < HH && w < HH) {
      float c = cosf((float)m * ((float)w + 0.5f) * (PI_F / (float)HH)) * sqrtf(2.f / (float)HH);
      if (m == 0) c *= 0.70710678118654752440f;
      v = c;
    }
    Dp[i] = f2bf(v);
  }
  if (i < HW) {
    int h = i / HH, w = i % HH;
    float k = PI_F / (float)HH;
    dfac[i] = k * k * (float)(h * h + w * w);   // wn^2 + wm^2 ; wexp^k = exp(-k*dfac)
  }
}

// ---------------- pack weight [N,K] fp32 -> WMMA B-fragment layout bf16 ----------------
// packed[(((kt*Nt16 + nt)*32 + lane)*16 + j] = w[nt*16 + (lane&15)][kt*32 + (lane>>4)*16 + j]
__global__ void pack_weight_kernel(const float* __restrict__ src,
                                   unsigned short* __restrict__ dst, int N, int K) {
  int p = blockIdx.x * blockDim.x + threadIdx.x;
  if (p >= N * K) return;
  int j  = p & 15;
  int L  = (p >> 4) & 31;
  int tl = p >> 9;
  int Nt = N >> 4;
  int nt = tl % Nt, kt = tl / Nt;
  int n = nt * 16 + (L & 15);
  int k = kt * 32 + (L >> 4) * 16 + j;
  dst[p] = f2bf(src[(size_t)n * K + k]);
}

// ---------------- depthwise 3x3 conv, NCHW fp32 -> NHWC bf16 ----------------
__global__ __launch_bounds__(256) void dwconv_kernel(const float* __restrict__ x,
                                                     const float* __restrict__ dw_w,
                                                     const float* __restrict__ dw_b,
                                                     unsigned short* __restrict__ y) {
  __shared__ unsigned short tile[HH * CC];
  const int t = threadIdx.x;
  const int b = blockIdx.x / HH, h = blockIdx.x % HH;
  const int wl = t & 63, cg = t >> 6;          // lanes over w, 4 channel groups
  if (wl < HH) {
    for (int c = cg; c < CC; c += 4) {
      const float* wp = dw_w + c * 9;
      float acc = dw_b[c];
      const float* xb = x + ((size_t)(b * CC + c) * HH) * HH;
      #pragma unroll
      for (int dh = -1; dh <= 1; ++dh) {
        int hh = h + dh;
        if (hh < 0 || hh >= HH) continue;
        const float* rowp = xb + hh * HH;
        #pragma unroll
        for (int dw = -1; dw <= 1; ++dw) {
          int ww = wl + dw;
          if (ww < 0 || ww >= HH) continue;
          acc += wp[(dh + 1) * 3 + (dw + 1)] * rowp[ww];
        }
      }
      tile[wl * CC + c] = f2bf(acc);
    }
  }
  __syncthreads();
  unsigned short* yo = y + (size_t)(b * HW + h * HH) * CC;
  for (int i = t; i < HH * CC; i += 256) yo[i] = tile[i];
}

// ---------------- generic WMMA GEMM: out = A[MxK](bf16) * Bp(packed) + bias ----------------
#define GM_BF16   0   // bf16 out, optional column split into out1
#define GM_DECAY  1   // out = exp(-relu(acc+bias)*dfac[m]) bf16
#define GM_F32    2   // fp32 out
#define GM_TSTORE 3   // fp32 transposed store: out[(b*192 + n)*3136 + m%3136]
#define PA 40

__global__ __launch_bounds__(128) void gemm_wmma_kernel(
    const unsigned short* __restrict__ A, const unsigned short* __restrict__ Bp,
    const float* __restrict__ bias, void* __restrict__ out0, void* __restrict__ out1,
    int M, int N, int K, int split, const float* __restrict__ dfac, int mode) {
  __shared__ unsigned short lA[2][64 * PA];   // double-buffered A tile
  const int t = threadIdx.x;
  const int wv = t >> 5, lane = t & 31, l = lane & 15, hi = lane >> 4;
  const int m0 = blockIdx.y * 64, n0 = blockIdx.x * 64;
  const int Ktiles = K >> 5, Nt16 = N >> 4;
  const int ntile = (n0 >> 4) + wv;

  FragC acc[4];
  #pragma unroll
  for (int s = 0; s < 4; ++s)
    #pragma unroll
    for (int j = 0; j < 8; ++j) acc[s].f[j] = 0.f;

  const int ar = t >> 1, ac = (t & 1) * 16;
  auto stageA = [&](int kt, int buf) {
    const unsigned short* gp = A + (size_t)(m0 + ar) * K + kt * 32 + ac;
    async_copy_b128(gp,     &lA[buf][ar * PA + ac]);
    async_copy_b128(gp + 8, &lA[buf][ar * PA + ac + 8]);
  };

  stageA(0, 0);
  async_wait_all();
  __syncthreads();

  for (int kt = 0; kt < Ktiles; ++kt) {
    const int buf = kt & 1;
    if (kt + 1 < Ktiles) stageA(kt + 1, buf ^ 1);   // overlap next tile with compute

    FragAB bf;   // B frag straight from packed global (coalesced 32B/lane, L2-resident)
    const unsigned short* bp = Bp + ((size_t)(kt * Nt16 + ntile) * 32 + lane) * 16;
    bf.q[0] = *(const U4*)bp;
    bf.q[1] = *(const U4*)(bp + 8);

    #pragma unroll
    for (int s = 0; s < 4; ++s) {
      FragAB af;
      const unsigned short* ap = &lA[buf][(s * 16 + l) * PA + hi * 16];
      af.q[0] = *(const U4*)ap;
      af.q[1] = *(const U4*)(ap + 8);
      acc[s].v = __builtin_amdgcn_wmma_f32_16x16x32_bf16(
          false, af.v, false, bf.v, (short)0, acc[s].v, false, false);
    }
    async_wait_all();
    __syncthreads();
  }

  const int n_g = n0 + wv * 16 + l;
  const float bn = bias ? bias[n_g] : 0.f;
  if (mode == GM_BF16) {
    #pragma unroll
    for (int s = 0; s < 4; ++s)
      #pragma unroll
      for (int v = 0; v < 8; ++v) {
        int m_g = m0 + s * 16 + v + 8 * hi;
        float val = acc[s].f[v] + bn;
        if (n_g < split) ((unsigned short*)out0)[(size_t)m_g * split + n_g] = f2bf(val);
        else ((unsigned short*)out1)[(size_t)m_g * (N - split) + (n_g - split)] = f2bf(val);
      }
  } else if (mode == GM_DECAY) {
    #pragma unroll
    for (int s = 0; s < 4; ++s)
      #pragma unroll
      for (int v = 0; v < 8; ++v) {
        int m_g = m0 + s * 16 + v + 8 * hi;
        float k = fmaxf(acc[s].f[v] + bn, 0.f);
        ((unsigned short*)out0)[(size_t)m_g * N + n_g] = f2bf(__expf(-k * dfac[m_g]));
      }
  } else if (mode == GM_F32) {
    #pragma unroll
    for (int s = 0; s < 4; ++s)
      #pragma unroll
      for (int v = 0; v < 8; ++v) {
        int m_g = m0 + s * 16 + v + 8 * hi;
        ((float*)out0)[(size_t)m_g * N + n_g] = acc[s].f[v] + bn;
      }
  } else { // GM_TSTORE  (N==192, rows never straddle a batch since 3136 % 64 == 0)
    #pragma unroll
    for (int s = 0; s < 4; ++s) {
      int mbase = m0 + s * 16 + 8 * hi;
      int b = mbase / HW, r = mbase - b * HW;
      float* op = (float*)out0 + ((size_t)b * CC + n_g) * HW + r;
      F4 v0 = { acc[s].f[0] + bn, acc[s].f[1] + bn, acc[s].f[2] + bn, acc[s].f[3] + bn };
      F4 v1 = { acc[s].f[4] + bn, acc[s].f[5] + bn, acc[s].f[6] + bn, acc[s].f[7] + bn };
      *(F4*)op = v0; *(F4*)(op + 4) = v1;
    }
  }
}

// ---------------- DCT apply: view in as [R,56,S]; out[r,m,s] = sum_w D[m,w] in[r,w,s] ----------------
#define PD 72
__global__ __launch_bounds__(128) void dct_wmma_kernel(
    const unsigned short* __restrict__ in, unsigned short* __restrict__ out,
    const unsigned short* __restrict__ Dp, int S) {
  __shared__ unsigned short lD[64 * PD];   // [m][w]
  __shared__ unsigned short lB[64 * PD];   // transposed: [s][w]
  const int t = threadIdx.x;
  const int wv = t >> 5, lane = t & 31, l = lane & 15, hi = lane >> 4;
  const int r = blockIdx.y, n0 = blockIdx.x * 64;

  { // stage D (64x64) asynchronously; overlaps with the transpose staging below
    int row = t >> 1, c0 = (t & 1) * 32;
    #pragma unroll
    for (int i = 0; i < 4; ++i)
      async_copy_b128(Dp + row * 64 + c0 + i * 8, &lD[row * PD + c0 + i * 8]);
  }
  #pragma unroll
  for (int it = 0; it < 4; ++it) { // stage B transposed (w rows 56..63 zeroed)
    int w = (t >> 3) + it * 16;
    int sc = (t & 7) * 8;
    U4 q = {0u, 0u, 0u, 0u};
    if (w < HH) q = *(const U4*)(in + (size_t)(r * HH + w) * S + n0 + sc);
    const unsigned short* hp = (const unsigned short*)&q;
    #pragma unroll
    for (int i = 0; i < 8; ++i) lB[(sc + i) * PD + w] = hp[i];
  }
  async_wait_all();
  __syncthreads();

  FragAB bfr[2];
  #pragma unroll
  for (int kc = 0; kc < 2; ++kc) {
    const unsigned short* bp = &lB[(wv * 16 + l) * PD + kc * 32 + hi * 16];
    bfr[kc].q[0] = *(const U4*)bp;
    bfr[kc].q[1] = *(const U4*)(bp + 8);
  }
  FragC acc[4];
  #pragma unroll
  for (int s = 0; s < 4; ++s)
    #pragma unroll
    for (int j = 0; j < 8; ++j) acc[s].f[j] = 0.f;

  #pragma unroll
  for (int s = 0; s < 4; ++s)
    #pragma unroll
    for (int kc = 0; kc < 2; ++kc) {
      FragAB af;
      const unsigned short* ap = &lD[(s * 16 + l) * PD + kc * 32 + hi * 16];
      af.q[0] = *(const U4*)ap;
      af.q[1] = *(const U4*)(ap + 8);
      acc[s].v = __builtin_amdgcn_wmma_f32_16x16x32_bf16(
          false, af.v, false, bfr[kc].v, (short)0, acc[s].v, false, false);
    }

  #pragma unroll
  for (int s = 0; s < 4; ++s)
    #pragma unroll
    for (int v = 0; v < 8; ++v) {
      int m = s * 16 + v + 8 * hi;
      if (m < HH)
        out[(size_t)(r * HH + m) * S + n0 + wv * 16 + l] = f2bf(acc[s].f[v]);
    }
}

// ---------------- small elementwise kernels ----------------
__global__ void feot_kernel(const float* __restrict__ freq, const float* __restrict__ outline,
                            unsigned short* __restrict__ dst) {
  int i = blockIdx.x * blockDim.x + threadIdx.x;
  if (i >= HW * CC) return;
  float s = 0.f;
  for (int b = 0; b < BB; ++b) {
    if (b + 4 < BB) __builtin_prefetch(&outline[(size_t)(b + 4) * (HW * CC) + i], 0, 1);
    s += outline[(size_t)b * (HW * CC) + i];
  }
  dst[i] = f2bf(freq[i] + s * (1.f / 64.f));
}

__global__ void xcmean_kernel(const unsigned short* __restrict__ xc, unsigned short* __restrict__ dst) {
  int i = blockIdx.x * blockDim.x + threadIdx.x;
  if (i >= HW * C2) return;
  float s = 0.f;
  for (int b = 0; b < BB; ++b) {
    if (b + 4 < BB) __builtin_prefetch(&xc[(size_t)(b + 4) * (HW * C2) + i], 0, 1);
    s += bf2f(xc[(size_t)b * (HW * C2) + i]);
  }
  dst[i] = f2bf(s * (1.f / 64.f));
}

__global__ void mulpw_kernel(unsigned short* __restrict__ x, const unsigned short* __restrict__ pw, int mod) {
  int i = blockIdx.x * blockDim.x + threadIdx.x;
  if (i >= mod) return;
  size_t idx = (size_t)blockIdx.y * mod + i;
  x[idx] = f2bf(bf2f(x[idx]) * bf2f(pw[i]));
}

__global__ void concat_kernel(const unsigned short* __restrict__ x1, const float* __restrict__ outline,
                              unsigned short* __restrict__ xc) {
  size_t m = blockIdx.x; int t = threadIdx.x;   // 384 threads
  xc[m * C2 + t] = (t < CC) ? x1[m * CC + t] : f2bf(outline[m * CC + (t - CC)]);
}

__global__ __launch_bounds__(256) void lnsilu_kernel(const float* __restrict__ u,
                                                     const unsigned short* __restrict__ z,
                                                     const float* __restrict__ g,
                                                     const float* __restrict__ bb,
                                                     unsigned short* __restrict__ v) {
  int wv = threadIdx.x >> 5, l = threadIdx.x & 31;
  size_t m = (size_t)blockIdx.x * 8 + wv;
  const float* up = u + m * CC;
  float vals[6], s = 0.f, s2 = 0.f;
  #pragma unroll
  for (int k = 0; k < 6; ++k) { float a = up[k * 32 + l]; vals[k] = a; s += a; s2 += a * a; }
  #pragma unroll
  for (int off = 16; off > 0; off >>= 1) { s += __shfl_xor(s, off, 32); s2 += __shfl_xor(s2, off, 32); }
  float mu = s * (1.f / 192.f);
  float var = s2 * (1.f / 192.f) - mu * mu;
  float rstd = rsqrtf(var + 1e-5f);
  #pragma unroll
  for (int k = 0; k < 6; ++k) {
    int c = k * 32 + l;
    float val = (vals[k] - mu) * rstd * g[c] + bb[c];
    float zf = bf2f(z[m * CC + c]);
    float sig = 1.f / (1.f + __expf(-zf));
    v[m * CC + c] = f2bf(val * zf * sig);
  }
}

// ---------------- workspace layout (all sizes already 256B multiples) ----------------
constexpr size_t OFF_DP    = 0;
constexpr size_t OFF_DFAC  = OFF_DP    + 64ull * 64 * 2;
constexpr size_t OFF_W1P   = OFF_DFAC  + 3136ull * 4;
constexpr size_t OFF_TOKP  = OFF_W1P   + 192ull * 384 * 2;
constexpr size_t OFF_W2P   = OFF_TOKP  + 192ull * 192 * 2;
constexpr size_t OFF_TOK2P = OFF_W2P   + 384ull * 384 * 2;
constexpr size_t OFF_W3P   = OFF_TOK2P + 384ull * 384 * 2;
constexpr size_t OFF_OUTWP = OFF_W3P   + 384ull * 192 * 2;
constexpr size_t OFF_FEOT  = OFF_OUTWP + 192ull * 192 * 2;
constexpr size_t OFF_PW1   = OFF_FEOT  + 3136ull * 192 * 2;
constexpr size_t OFF_PW2   = OFF_PW1   + 3136ull * 192 * 2;
constexpr size_t OFF_SKF   = OFF_PW2   + 3136ull * 384 * 2;
constexpr size_t OFF_XCM   = OFF_SKF   + 3136ull * 384 * 2;
constexpr size_t OFF_Y     = OFF_XCM   + 3136ull * 384 * 2;          // y, then t1, then v
constexpr size_t OFF_X1    = OFF_Y     + 200704ull * 192 * 2;
constexpr size_t OFF_Z     = OFF_X1    + 200704ull * 192 * 2;
constexpr size_t OFF_XC    = OFF_Z     + 200704ull * 192 * 2;
constexpr size_t OFF_T2    = OFF_XC    + 200704ull * 384 * 2;        // t2 (bf16) / u (fp32)
constexpr size_t WS_NEED   = OFF_T2    + 200704ull * 384 * 2;        // ~525 MiB

extern "C" void kernel_launch(void* const* d_in, const int* in_sizes, int n_in,
                              void* d_out, int out_size, void* d_ws, size_t ws_size,
                              hipStream_t stream) {
  (void)in_sizes; (void)n_in; (void)out_size;
  if (ws_size < WS_NEED) return;

  const float* x        = (const float*)d_in[0];
  const float* freq     = (const float*)d_in[1];
  const float* outline  = (const float*)d_in[2];
  const float* dw_w     = (const float*)d_in[3];
  const float* dw_b     = (const float*)d_in[4];
  const float* lin_w    = (const float*)d_in[5];
  const float* lin_b    = (const float*)d_in[6];
  const float* tok_w    = (const float*)d_in[7];
  const float* tok_b    = (const float*)d_in[8];
  const float* lin2_w   = (const float*)d_in[9];
  const float* lin2_b   = (const float*)d_in[10];
  const float* tok2_w   = (const float*)d_in[11];
  const float* tok2_b   = (const float*)d_in[12];
  const float* lin3_w   = (const float*)d_in[13];
  const float* lin3_b   = (const float*)d_in[14];
  const float* norm_g   = (const float*)d_in[15];
  const float* norm_b   = (const float*)d_in[16];
  const float* out_w    = (const float*)d_in[17];
  const float* out_b    = (const float*)d_in[18];

  char* ws = (char*)d_ws;
  unsigned short* Dp   = (unsigned short*)(ws + OFF_DP);
  float*          dfac = (float*)(ws + OFF_DFAC);
  unsigned short* w1p  = (unsigned short*)(ws + OFF_W1P);
  unsigned short* tokp = (unsigned short*)(ws + OFF_TOKP);
  unsigned short* w2p  = (unsigned short*)(ws + OFF_W2P);
  unsigned short* tk2p = (unsigned short*)(ws + OFF_TOK2P);
  unsigned short* w3p  = (unsigned short*)(ws + OFF_W3P);
  unsigned short* owp  = (unsigned short*)(ws + OFF_OUTWP);
  unsigned short* feot = (unsigned short*)(ws + OFF_FEOT);
  unsigned short* pw1  = (unsigned short*)(ws + OFF_PW1);
  unsigned short* pw2  = (unsigned short*)(ws + OFF_PW2);
  unsigned short* skf  = (unsigned short*)(ws + OFF_SKF);
  unsigned short* xcm  = (unsigned short*)(ws + OFF_XCM);
  unsigned short* y    = (unsigned short*)(ws + OFF_Y);     // reused: t1, v
  unsigned short* x1   = (unsigned short*)(ws + OFF_X1);
  unsigned short* z    = (unsigned short*)(ws + OFF_Z);
  unsigned short* xc   = (unsigned short*)(ws + OFF_XC);
  unsigned short* t2   = (unsigned short*)(ws + OFF_T2);
  float*          u    = (float*)(ws + OFF_T2);             // fp32 alias after t2 retired
  unsigned short* t1   = y;
  unsigned short* v    = y;

  auto gemm = [&](const unsigned short* A, const unsigned short* Bp, const float* bias,
                  void* o0, void* o1, int M, int N, int K, int split, const float* df, int mode) {
    gemm_wmma_kernel<<<dim3(N / 64, M / 64), 128, 0, stream>>>(A, Bp, bias, o0, o1,
                                                               M, N, K, split, df, mode);
  };
  auto dct = [&](const unsigned short* in, unsigned short* out, int R, int S) {
    dct_wmma_kernel<<<dim3(S / 64, R), 128, 0, stream>>>(in, out, Dp, S);
  };

  // constants + packed weights
  prep_const_kernel<<<16, 256, 0, stream>>>(Dp, dfac);
  pack_weight_kernel<<<(384 * 192 + 255) / 256, 256, 0, stream>>>(lin_w,  w1p, 384, 192);
  pack_weight_kernel<<<(192 * 192 + 255) / 256, 256, 0, stream>>>(tok_w,  tokp, 192, 192);
  pack_weight_kernel<<<(384 * 384 + 255) / 256, 256, 0, stream>>>(lin2_w, w2p, 384, 384);
  pack_weight_kernel<<<(384 * 384 + 255) / 256, 256, 0, stream>>>(tok2_w, tk2p, 384, 384);
  pack_weight_kernel<<<(192 * 384 + 255) / 256, 256, 0, stream>>>(lin3_w, w3p, 192, 384);
  pack_weight_kernel<<<(192 * 192 + 255) / 256, 256, 0, stream>>>(out_w,  owp, 192, 192);

  // k1 path: pw1 = wexp ^ relu((freq + mean_b(outline)) @ tok^T + b)
  feot_kernel<<<(HW * CC + 255) / 256, 256, 0, stream>>>(freq, outline, feot);
  gemm(feot, tokp, tok_b, pw1, nullptr, HW, 192, 192, 192, dfac, GM_DECAY);

  // main: dwconv -> lin (split x/z) -> DCT/DCT -> *pw1 -> DCT/DCT
  dwconv_kernel<<<BB * HH, 256, 0, stream>>>(x, dw_w, dw_b, y);
  gemm(y, w1p, lin_b, x1, z, MM, 384, 192, 192, nullptr, GM_BF16);
  dct(x1, t1, BB, HH * CC);                 // H transform
  dct(t1, x1, BB * HH, CC);                 // W transform
  mulpw_kernel<<<dim3(HW * CC / 256, BB), 256, 0, stream>>>(x1, pw1, HW * CC);
  dct(x1, t1, BB, HH * CC);                 // "inverse" H (same matrix per einsum semantics)
  dct(t1, x1, BB * HH, CC);                 // "inverse" W

  // concat + batch-mean-folded lin2/tok2 -> pw2
  concat_kernel<<<MM, 384, 0, stream>>>(x1, outline, xc);
  xcmean_kernel<<<(HW * C2 + 255) / 256, 256, 0, stream>>>(xc, xcm);
  gemm(xcm, w2p, lin2_b, skf, nullptr, HW, 384, 384, 384, nullptr, GM_BF16);
  gemm(skf, tk2p, tok2_b, pw2, nullptr, HW, 384, 384, 384, dfac, GM_DECAY);

  // second DCT pass on 2C (reference's interleaved reshape == contiguous [B*H*2,56,192])
  dct(xc, t2, BB, HH * C2);
  dct(t2, xc, BB * HH * 2, CC);
  mulpw_kernel<<<dim3(HW * C2 / 256, BB), 256, 0, stream>>>(xc, pw2, HW * C2);
  dct(xc, t2, BB, HH * C2);
  dct(t2, xc, BB * HH * 2, CC);

  // lin3 -> LayerNorm -> *silu(z) -> out proj (transposed store to NCHW)
  gemm(xc, w3p, lin3_b, u, nullptr, MM, 192, 384, 192, nullptr, GM_F32);
  lnsilu_kernel<<<MM / 8, 256, 0, stream>>>(u, z, norm_g, norm_b, v);
  gemm(v, owp, out_b, d_out, nullptr, MM, 192, 192, 192, nullptr, GM_TSTORE);
}
